// Seq2SeqLSTM_82317343195876
// MI455X (gfx1250) — matrix operand order repaired and checked by reading, not compiled, not run on hardware
//
#include <hip/hip_runtime.h>
#include <hip/hip_bf16.h>
#include <stdint.h>

// ---------------- problem constants (from reference) ----------------
#define T_IN   336
#define T_OUT  96
#define BATCH  2048
#define DIN    64            // input_size == output_size
#define HID    512
#define G4     (4 * HID)     // 2048 gate rows of Wcat
#define KCAT   (DIN + HID)   // 576: concatenated K = [x | h]
#define NKC    (KCAT / 32)   // 18 K-chunks of 32

// ---------------- WMMA types ----------------
typedef __attribute__((ext_vector_type(16))) __bf16       v16bf;
typedef __attribute__((ext_vector_type(8)))  float        v8f;
typedef __attribute__((ext_vector_type(4)))  unsigned int u32x4;

struct P128x2 { u32x4 lo; u32x4 hi; };
static __device__ __forceinline__ v16bf make_frag(u32x4 lo, u32x4 hi) {
  P128x2 p{lo, hi};
  return __builtin_bit_cast(v16bf, p);
}

static __device__ __forceinline__ unsigned short f2bf(float f) {
  unsigned u = __float_as_uint(f);
  u += 0x7FFFu + ((u >> 16) & 1u);
  return (unsigned short)(u >> 16);
}
static __device__ __forceinline__ float sigmoidf_(float x) {
  return 1.0f / (1.0f + __expf(-x));
}

// ---------------- CDNA5 async global -> LDS copy (ASYNCcnt-tracked) ----------
// Per-lane 16B transfer: global bytes at vaddr+ioff go to LDS at vdst+ioff.
// Immediate byte offset keeps per-chunk address math out of the VALU
// (loop is fully unrolled, so "i" folds to the instruction's offset field).
static __device__ __forceinline__ void async_copy_b128(unsigned lds_off,
                                                       const unsigned short* g,
                                                       int ioff_bytes) {
  asm volatile("global_load_async_to_lds_b128 %0, %1, off offset:%2"
               :: "v"(lds_off), "v"(g), "i"(ioff_bytes)
               : "memory");
}
static __device__ __forceinline__ void wait_async_le4() {
  asm volatile("s_wait_asynccnt 0x4" ::: "memory");
}
static __device__ __forceinline__ void wait_async_le2() {
  asm volatile("s_wait_asynccnt 0x2" ::: "memory");
}
static __device__ __forceinline__ void wait_async_le0() {
  asm volatile("s_wait_asynccnt 0x0" ::: "memory");
}
// low 32 bits of a generic LDS pointer == LDS byte address (aperture rule)
static __device__ __forceinline__ unsigned lds_off32(const void* p) {
  return (unsigned)(uintptr_t)p;
}

// ---------------- one-time packing kernels ----------------
__global__ __launch_bounds__(256) void pack_weights(
    const float* __restrict__ Wih,        // [G4, DIN]
    const float* __restrict__ Whh,        // [G4, HID]
    unsigned short* __restrict__ Wcat)    // [G4, KCAT] bf16 bits
{
  int idx = blockIdx.x * blockDim.x + threadIdx.x;
  if (idx >= G4 * KCAT) return;
  int n = idx / KCAT;
  int k = idx - n * KCAT;
  float v = (k < DIN) ? Wih[(size_t)n * DIN + k] : Whh[(size_t)n * HID + (k - DIN)];
  Wcat[idx] = f2bf(v);
}

__global__ __launch_bounds__(256) void pack_bias(
    const float* __restrict__ bih, const float* __restrict__ bhh,
    float* __restrict__ bsum)             // [G4]
{
  int i = blockIdx.x * blockDim.x + threadIdx.x;
  if (i < G4) bsum[i] = bih[i] + bhh[i];
}

// ---------------- stage x_t (f32 -> bf16) into A[:, 0:DIN] ----------------
__global__ __launch_bounds__(256) void pack_x(
    const float* __restrict__ x,          // [BATCH, DIN] slice
    unsigned short* __restrict__ A)       // [BATCH, KCAT]
{
  int idx = blockIdx.x * blockDim.x + threadIdx.x;   // BATCH*DIN threads
  int b = idx / DIN;
  int d = idx - b * DIN;
  A[(size_t)b * KCAT + d] = f2bf(x[idx]);
}

__global__ __launch_bounds__(256) void zero_x(unsigned short* __restrict__ A) {
  int idx = blockIdx.x * blockDim.x + threadIdx.x;
  int b = idx / DIN;
  int d = idx - b * DIN;
  A[(size_t)b * KCAT + d] = 0;                       // bf16 +0.0
}

// ---------------- fused LSTM step: gate GEMM (bf16 WMMA) + cell pointwise ----
// Grid: (BATCH/64) x (HID/16). Block tile = 64 batch rows x 64 gate cols where
// the 64 cols are the four gate segments {g*HID + j0 + 0..15 : g=0..3} of one
// 16-wide j-tile, so the block holds i,f,g,o for its (row, j) patch and does
// the sigmoid/tanh cell update in-kernel (no gates round-trip to HBM).
//
// K-loop fully unrolled; LDS uses a 4-deep buffer ring with distance-2 async
// prefetch (global_load_async_to_lds_b128). Ring depth 4 makes the prefetch
// target disjoint from the buffers any wave (up to one iteration behind) may
// still be reading, so ONE barrier per chunk suffices. Async loads complete
// in order -> exact s_wait_asynccnt counts (0x4 steady / 0x2 / 0x0 tail).
__global__ __launch_bounds__(256) void lstm_step_fused(
    const unsigned short* __restrict__ Ain,   // [BATCH, KCAT] bf16 (x_t | h_{t-1})
    const unsigned short* __restrict__ Wcat,  // [G4,   KCAT] bf16
    const float* __restrict__ bsum,           // [G4] = bih + bhh
    float* __restrict__ c,                    // [BATCH, HID] in/out
    float* __restrict__ h,                    // [BATCH, HID] out (f32 for projection)
    unsigned short* __restrict__ Aout)        // [BATCH, KCAT]; h part written here
{
  // 40-elem row pad: 80B stride (multiple of 16B) -> b128-aligned fragments
  __shared__ __align__(16) unsigned short Abuf[4][64][40];
  __shared__ __align__(16) unsigned short Wbuf[4][64][40];
  __shared__ float Gs[64][68];               // gate exchange tile (f32)

  const int tid  = threadIdx.x;
  const int wave = tid >> 5;
  const int lane = tid & 31;
  const int bm = blockIdx.x * 64;            // batch-row base
  const int j0 = blockIdx.y * 16;            // hidden-column base

  // cooperative staging: each thread owns one 16B lane-transfer per tile
  const int lr = tid >> 2;                   // 0..63
  const int lc = (tid & 3) * 8;              // 0,8,16,24
  const int wseg = lr >> 4;                  // gate segment of this W row
  const int wrow = wseg * HID + j0 + (lr & 15);

  const unsigned short* gA = Ain  + (size_t)(bm + lr) * KCAT + lc;
  const unsigned short* gW = Wcat + (size_t)wrow * KCAT + lc;
  unsigned ldsA[4], ldsW[4];
#pragma unroll
  for (int s = 0; s < 4; ++s) {
    ldsA[s] = lds_off32(&Abuf[s][lr][lc]);
    ldsW[s] = lds_off32(&Wbuf[s][lr][lc]);
  }

  const int half = lane >> 4;                // bf16 WMMA K split across lane halves
  const int wm = (wave & 3) * 16;
  const int wn = (wave >> 2) * 32;
  const int mrow = wm + (lane & 15);
  const int nc0  = wn + (lane & 15);
  const int nc1  = nc0 + 16;

  // hoist c-state loads: overlap their HBM latency with the whole GEMM
  float cold[4];
#pragma unroll
  for (int q = 0; q < 4; ++q) {
    const int e = tid + q * 256;
    cold[q] = c[(size_t)(bm + (e >> 4)) * HID + j0 + (e & 15)];
  }

  v8f acc0 = {}; v8f acc1 = {};

  // prologue: async-stage chunks 0 and 1 (distance-2 pipeline)
  async_copy_b128(ldsA[0], gA, 0);
  async_copy_b128(ldsW[0], gW, 0);
  async_copy_b128(ldsA[1], gA, 64);
  async_copy_b128(ldsW[1], gW, 64);

#pragma unroll
  for (int kc = 0; kc < NKC; ++kc) {
    const int cur = kc & 3;
    if (kc + 2 < NKC) {                      // prefetch chunk kc+2
      const int nxt = (kc + 2) & 3;
      async_copy_b128(ldsA[nxt], gA, (kc + 2) * 64);
      async_copy_b128(ldsW[nxt], gW, (kc + 2) * 64);
      wait_async_le4();                      // chunks kc+1,kc+2 in flight; kc landed
    } else if (kc + 1 < NKC) {
      wait_async_le2();                      // only chunk kc+1 in flight
    } else {
      wait_async_le0();
    }
    __syncthreads();                         // chunk kc resident on all waves

    // A 16x32 bf16: lanes 0-15 K=0..15, lanes 16-31 K=16..31
    v16bf a  = make_frag(*(const u32x4*)&Abuf[cur][mrow][half * 16],
                         *(const u32x4*)&Abuf[cur][mrow][half * 16 + 8]);
    // B 32x16 bf16: lane&15 = column; lanes 0-15 K=0..15, lanes 16-31 K=16..31
    v16bf b0 = make_frag(*(const u32x4*)&Wbuf[cur][nc0][half * 16],
                         *(const u32x4*)&Wbuf[cur][nc0][half * 16 + 8]);
    v16bf b1 = make_frag(*(const u32x4*)&Wbuf[cur][nc1][half * 16],
                         *(const u32x4*)&Wbuf[cur][nc1][half * 16 + 8]);

    acc0 = __builtin_amdgcn_wmma_f32_16x16x32_bf16(false, a, false, b0,
                                                   (short)0, acc0, false, false);
    acc1 = __builtin_amdgcn_wmma_f32_16x16x32_bf16(false, a, false, b1,
                                                   (short)0, acc1, false, false);
    // no bottom barrier: ring depth 4 keeps prefetch target disjoint from
    // any buffer still being read by a wave at most one iteration behind
  }

  // spill accumulators to LDS for cross-wave gate exchange
  {
    const int r  = wm + half * 8;
    const int n0 = wn + (lane & 15);
#pragma unroll
    for (int v = 0; v < 8; ++v) {
      Gs[r + v][n0]      = acc0[v];
      Gs[r + v][n0 + 16] = acc1[v];
    }
  }
  __syncthreads();

  // pointwise LSTM cell: 64 rows x 16 j = 1024 elements, 4 per thread
#pragma unroll
  for (int q = 0; q < 4; ++q) {
    const int e   = tid + q * 256;
    const int r   = e >> 4;
    const int jj  = e & 15;
    const int j   = j0 + jj;
    const int row = bm + r;
    // PyTorch gate order i, f, g, o along 4H
    float gi = Gs[r][jj]      + bsum[j];
    float gf = Gs[r][16 + jj] + bsum[HID + j];
    float gg = Gs[r][32 + jj] + bsum[2 * HID + j];
    float go = Gs[r][48 + jj] + bsum[3 * HID + j];
    const size_t ci = (size_t)row * HID + j;
    float cn = sigmoidf_(gf) * cold[q] + sigmoidf_(gi) * tanhf(gg);
    float hn = sigmoidf_(go) * tanhf(cn);
    c[ci] = cn;
    h[ci] = hn;
    Aout[(size_t)row * KCAT + DIN + j] = f2bf(hn);   // next-step recurrent input
  }
}

// ---------------- decoder projection: y = h @ lin_WT + lin_b; feedback ----------
__global__ __launch_bounds__(256) void dec_linear(
    const float* __restrict__ h,        // [BATCH, HID]
    const float* __restrict__ lin_W,    // [DIN, HID]
    const float* __restrict__ lin_b,    // [DIN]
    float* __restrict__ y_out,          // d_out + t*BATCH*DIN
    unsigned short* __restrict__ Anext) // [BATCH, KCAT]; x part of next dec input
{
  int idx = blockIdx.x * blockDim.x + threadIdx.x;  // BATCH*DIN threads
  int b = idx / DIN;
  int d = idx - b * DIN;
  const float* hb = h + (size_t)b * HID;
  const float* wr = lin_W + (size_t)d * HID;
  float acc = lin_b[d];
#pragma unroll 8
  for (int k = 0; k < HID; ++k) acc = fmaf(hb[k], wr[k], acc);
  y_out[idx] = acc;                                  // output at time t
  Anext[(size_t)b * KCAT + d] = f2bf(acc);           // detached feedback input
}

// ---------------- host-side orchestration (graph-capture safe) ----------------
extern "C" void kernel_launch(void* const* d_in, const int* in_sizes, int n_in,
                              void* d_out, int out_size, void* d_ws, size_t ws_size,
                              hipStream_t stream) {
  (void)in_sizes; (void)n_in; (void)out_size; (void)ws_size;
  const float* inputs   = (const float*)d_in[0];   // [T_IN, BATCH, DIN]
  const float* enc_Wih  = (const float*)d_in[1];
  const float* enc_Whh  = (const float*)d_in[2];
  const float* enc_bih  = (const float*)d_in[3];
  const float* enc_bhh  = (const float*)d_in[4];
  const float* dec_Wih  = (const float*)d_in[5];
  const float* dec_Whh  = (const float*)d_in[6];
  const float* dec_bih  = (const float*)d_in[7];
  const float* dec_bhh  = (const float*)d_in[8];
  const float* lin_W    = (const float*)d_in[9];
  const float* lin_b    = (const float*)d_in[10];
  float* out = (float*)d_out;

  // workspace carve-up (256B aligned regions)
  uint8_t* wp = (uint8_t*)d_ws;
  auto take = [&](size_t bytes) -> void* {
    void* p = (void*)wp;
    wp += (bytes + 255) & ~(size_t)255;
    return p;
  };
  unsigned short* WcatE = (unsigned short*)take((size_t)G4 * KCAT * 2);
  unsigned short* WcatD = (unsigned short*)take((size_t)G4 * KCAT * 2);
  float* bsumE          = (float*)take((size_t)G4 * 4);
  float* bsumD          = (float*)take((size_t)G4 * 4);
  unsigned short* A0    = (unsigned short*)take((size_t)BATCH * KCAT * 2);
  unsigned short* A1    = (unsigned short*)take((size_t)BATCH * KCAT * 2);
  float* hbuf           = (float*)take((size_t)BATCH * HID * 4);
  float* cbuf           = (float*)take((size_t)BATCH * HID * 4);
  unsigned short* A[2]  = { A0, A1 };

  // fresh state every call (deterministic, replay-safe)
  hipMemsetAsync(A0,   0, (size_t)BATCH * KCAT * 2, stream);
  hipMemsetAsync(A1,   0, (size_t)BATCH * KCAT * 2, stream);
  hipMemsetAsync(cbuf, 0, (size_t)BATCH * HID * 4, stream);

  const int wblk = (G4 * KCAT + 255) / 256;
  pack_weights<<<wblk, 256, 0, stream>>>(enc_Wih, enc_Whh, WcatE);
  pack_weights<<<wblk, 256, 0, stream>>>(dec_Wih, dec_Whh, WcatD);
  pack_bias<<<(G4 + 255) / 256, 256, 0, stream>>>(enc_bih, enc_bhh, bsumE);
  pack_bias<<<(G4 + 255) / 256, 256, 0, stream>>>(dec_bih, dec_bhh, bsumD);

  dim3 sgrid(BATCH / 64, HID / 16);            // 32 x 32 workgroups
  const int xblk = (BATCH * DIN) / 256;

  // ---- encoder: 336 sequential steps, ping-pong A buffers ----
  pack_x<<<xblk, 256, 0, stream>>>(inputs, A[0]);
  for (int t = 0; t < T_IN; ++t) {
    const int p = t & 1;
    lstm_step_fused<<<sgrid, 256, 0, stream>>>(A[p], WcatE, bsumE,
                                               cbuf, hbuf, A[p ^ 1]);
    if (t + 1 < T_IN)
      pack_x<<<xblk, 256, 0, stream>>>(inputs + (size_t)(t + 1) * BATCH * DIN,
                                       A[p ^ 1]);
  }

  // ---- decoder: x0 = zeros, then autoregressive feedback ----
  zero_x<<<xblk, 256, 0, stream>>>(A[T_IN & 1]);
  for (int t = 0; t < T_OUT; ++t) {
    const int p = (T_IN + t) & 1;
    lstm_step_fused<<<sgrid, 256, 0, stream>>>(A[p], WcatD, bsumD,
                                               cbuf, hbuf, A[p ^ 1]);
    dec_linear<<<xblk, 256, 0, stream>>>(hbuf, lin_W, lin_b,
                                         out + (size_t)t * BATCH * DIN, A[p ^ 1]);
  }
}
